// GNN_4655744549602
// MI455X (gfx1250) — matrix operand order repaired and checked by reading, not compiled, not run on hardware
//
#include <hip/hip_runtime.h>
#include <hip/hip_bf16.h>
#include <stdint.h>

#define N_   40000
#define E_   120000
#define G_   512
#define D_   512
#define DIN_ 128
#define EIN_ 16
#define L_   5
#define T_   128

typedef __attribute__((ext_vector_type(16))) __bf16    bf16x16;
typedef __attribute__((ext_vector_type(8)))  float     f32x8;
typedef __attribute__((ext_vector_type(4)))  uint32_t  u32x4;   // POD 16-byte vector

__device__ __forceinline__ uint16_t f2bf(float f) {
  uint32_t u = __float_as_uint(f);
  uint32_t r = u + 0x7FFFu + ((u >> 16) & 1u);
  return (uint16_t)(r >> 16);
}

// Force the global (AS1) load path: generic->global addrspace cast (device only).
__device__ __forceinline__ u32x4 ldg4(const u32x4* p) {
#if defined(__HIP_DEVICE_COMPILE__)
  return *(const __attribute__((address_space(1))) u32x4*)p;
#else
  return *p;
#endif
}

// ---------------------------------------------------------------------------
// bf16 WMMA GEMM: C[M,N] = A[M,K](bf16 row-major) @ B + bias
// B is pre-packed fragment-contiguous (see gnn_pack_w):
//   dword[((ct*KS + ks)*32 + lane)*8 + v] = bf16(B[k,c]) | bf16(B[k+1,c])<<16
//   with c = ct*16 + (lane&15), k = ks*32 + (lane>>4)*16 + 2v, KS = K/32.
// block = (32,2): 2 waves, each wave computes a 32x64 output strip
// (2 row-tiles x 4 col-tiles = 8 WMMAs per 32-K step, 2 A frags + 4 B frags).
// Requires: M % 64 == 0, K % 32 == 0, N % 64 == 0.
// ---------------------------------------------------------------------------
union AFrag { bf16x16 v; uint32_t u[8]; u32x4 q[2]; };

__global__ __launch_bounds__(64)
void gnn_wmma_gemm(const uint16_t* __restrict__ A,
                   const uint32_t* __restrict__ Bt,
                   const float* __restrict__ bias,
                   float* __restrict__ C,
                   int M, int K, int N)
{
  const int lane = threadIdx.x;          // 0..31
  const int half = lane >> 4;            // 0/1
  const int lr   = lane & 15;
  const int row0 = (blockIdx.y * 2 + threadIdx.y) << 5;   // 32 rows per wave
  const int col0 = blockIdx.x << 6;
  const int KS   = K >> 5;               // 32-wide K-steps

  f32x8 acc0[4], acc1[4];
  #pragma unroll
  for (int t = 0; t < 4; ++t) {
    acc0[t] = (f32x8){0.f,0.f,0.f,0.f,0.f,0.f,0.f,0.f};
    acc1[t] = (f32x8){0.f,0.f,0.f,0.f,0.f,0.f,0.f,0.f};
  }

  // A: row-major bf16; fragment rows row0+lr and row0+16+lr.
  const u32x4* Ap = (const u32x4*)((const uint32_t*)A + (((size_t)(row0 + lr) * (size_t)K) >> 1));
  const int arow16 = K << 1;             // 16 rows = 16*K bf16 = 2*K u32x4

  // B: fragment-contiguous; scalar pointers (no array) so AS inference works.
  const size_t tstride = (size_t)KS << 6;                 // u32x4 per col-tile
  const u32x4* B0 = (const u32x4*)Bt + ((size_t)(col0 >> 4) * (size_t)KS) * 64 + (lane << 1);
  const u32x4* B1 = B0 + tstride;
  const u32x4* B2 = B1 + tstride;
  const u32x4* B3 = B2 + tstride;

  for (int ks = 0; ks < KS; ++ks) {
    AFrag a0, a1;
    a0.q[0] = ldg4(Ap + 2 * half);
    a0.q[1] = ldg4(Ap + 2 * half + 1);
    a1.q[0] = ldg4(Ap + arow16 + 2 * half);
    a1.q[1] = ldg4(Ap + arow16 + 2 * half + 1);
    Ap += 4;

    AFrag b;
    // tile 0
    b.q[0] = ldg4(B0); b.q[1] = ldg4(B0 + 1); B0 += 64;
    acc0[0] = __builtin_amdgcn_wmma_f32_16x16x32_bf16(false, a0.v, false, b.v, (short)0, acc0[0], false, false);
    acc1[0] = __builtin_amdgcn_wmma_f32_16x16x32_bf16(false, a1.v, false, b.v, (short)0, acc1[0], false, false);
    // tile 1
    b.q[0] = ldg4(B1); b.q[1] = ldg4(B1 + 1); B1 += 64;
    acc0[1] = __builtin_amdgcn_wmma_f32_16x16x32_bf16(false, a0.v, false, b.v, (short)0, acc0[1], false, false);
    acc1[1] = __builtin_amdgcn_wmma_f32_16x16x32_bf16(false, a1.v, false, b.v, (short)0, acc1[1], false, false);
    // tile 2
    b.q[0] = ldg4(B2); b.q[1] = ldg4(B2 + 1); B2 += 64;
    acc0[2] = __builtin_amdgcn_wmma_f32_16x16x32_bf16(false, a0.v, false, b.v, (short)0, acc0[2], false, false);
    acc1[2] = __builtin_amdgcn_wmma_f32_16x16x32_bf16(false, a1.v, false, b.v, (short)0, acc1[2], false, false);
    // tile 3
    b.q[0] = ldg4(B3); b.q[1] = ldg4(B3 + 1); B3 += 64;
    acc0[3] = __builtin_amdgcn_wmma_f32_16x16x32_bf16(false, a0.v, false, b.v, (short)0, acc0[3], false, false);
    acc1[3] = __builtin_amdgcn_wmma_f32_16x16x32_bf16(false, a1.v, false, b.v, (short)0, acc1[3], false, false);
  }

  // C/D 16x16 f32 layout: VGPR r -> row = rowTile + half*8 + r, col = col0+16t+lr
  const int cbase = col0 + lr;
  const int idx0  = (row0 + (half << 3)) * N + cbase;     // fits in 32 bits
  #pragma unroll
  for (int t = 0; t < 4; ++t) {
    const float bv = bias ? bias[cbase + (t << 4)] : 0.0f;
    #pragma unroll
    for (int r = 0; r < 8; ++r) {
      C[(uint32_t)(idx0 + r * N + (t << 4))] = acc0[t][r] + bv;
      C[(uint32_t)(idx0 + (r + 16) * N + (t << 4))] = acc1[t][r] + bv;
    }
  }
}

// ---------------------------------------------------------------------------
// Weight pack: f32 [K,N] -> fragment-contiguous paired bf16 dwords (see above)
// ---------------------------------------------------------------------------
__global__ void gnn_pack_w(const float* __restrict__ W, uint32_t* __restrict__ out,
                           int K, int N)
{
  size_t i = (size_t)blockIdx.x * blockDim.x + threadIdx.x;
  size_t total = ((size_t)(K >> 1)) * (size_t)N;
  if (i >= total) return;
  const int KS = K >> 5;
  const int v    = (int)(i & 7);
  const int lane = (int)((i >> 3) & 31);
  const size_t rest = i >> 8;
  const int ks = (int)(rest % (size_t)KS);
  const int ct = (int)(rest / (size_t)KS);
  const int col = (ct << 4) + (lane & 15);
  const int k   = (ks << 5) + ((lane >> 4) << 4) + (v << 1);
  uint32_t lo = f2bf(W[(size_t)k * N + col]);
  uint32_t hi = f2bf(W[(size_t)(k + 1) * N + col]);
  out[i] = lo | (hi << 16);
}

// f32 -> bf16 row-major (2 elems/thread, packed dword store)
__global__ void gnn_cvt_bf16(const float* __restrict__ in, uint16_t* __restrict__ out,
                             size_t n2)
{
  size_t i = (size_t)blockIdx.x * blockDim.x + threadIdx.x;
  if (i >= n2) return;
  uint32_t lo = f2bf(in[2 * i]);
  uint32_t hi = f2bf(in[2 * i + 1]);
  ((uint32_t*)out)[i] = lo | (hi << 16);
}

__global__ void gnn_zero_f32(float* __restrict__ p, size_t n)
{
  size_t i = (size_t)blockIdx.x * blockDim.x + threadIdx.x;
  if (i < n) p[i] = 0.0f;
}

__global__ void gnn_copy_f32(float* __restrict__ dst, const float* __restrict__ src, size_t n)
{
  size_t i = (size_t)blockIdx.x * blockDim.x + threadIdx.x;
  if (i < n) dst[i] = src[i];
}

__global__ void gnn_vn_init(float* __restrict__ vn, const float* __restrict__ emb, int n)
{
  int i = blockIdx.x * blockDim.x + threadIdx.x;
  if (i < n) vn[i] = emb[i & (D_ - 1)];
}

// h += vn[batch]; zin = (1+eps)*h; optional pooled += h   (4 elems/thread)
__global__ void gnn_add_vn(float* __restrict__ h, const float* __restrict__ vn,
                           const int* __restrict__ batch, const float* __restrict__ epsp,
                           float* __restrict__ zin, float* __restrict__ pooled, int doPool)
{
  size_t g = (size_t)blockIdx.x * blockDim.x + threadIdx.x;
  if (g >= (size_t)N_ * (D_ / 4)) return;
  const int row = (int)(g >> 7);            // D/4 = 128
  const int c4  = ((int)g & 127) << 2;
  const int b   = batch[row];
  const float ep = 1.0f + *epsp;
  const float4 hv = *(const float4*)(h  + (size_t)row * D_ + c4);
  const float4 vv = *(const float4*)(vn + (size_t)b   * D_ + c4);
  float4 nh;
  nh.x = hv.x + vv.x; nh.y = hv.y + vv.y; nh.z = hv.z + vv.z; nh.w = hv.w + vv.w;
  *(float4*)(h + (size_t)row * D_ + c4) = nh;
  float4 z;
  z.x = ep * nh.x; z.y = ep * nh.y; z.z = ep * nh.z; z.w = ep * nh.w;
  *(float4*)(zin + (size_t)row * D_ + c4) = z;
  if (doPool) {
    float* pp = pooled + (size_t)b * D_ + c4;
    atomicAdd(pp + 0, nh.x); atomicAdd(pp + 1, nh.y);
    atomicAdd(pp + 2, nh.z); atomicAdd(pp + 3, nh.w);
  }
}

// Per-edge: emb = edge_attr[e] @ eW + eb (K=16); msg = relu(h[src]+emb); zin[dst] += msg
__global__ __launch_bounds__(128)
void gnn_edge_msg(const float* __restrict__ h, const float* __restrict__ ea,
                  const int* __restrict__ src, const int* __restrict__ dst,
                  const float* __restrict__ eW, const float* __restrict__ eb,
                  float* __restrict__ zin)
{
  const int e = blockIdx.x;
  __shared__ float a[EIN_];
  if (threadIdx.x < EIN_) a[threadIdx.x] = ea[(size_t)e * EIN_ + threadIdx.x];
  __syncthreads();
  const int s = src[e], d = dst[e];
  const int c = threadIdx.x << 2;           // 0..508
  float4 acc = *(const float4*)(eb + c);
  #pragma unroll
  for (int k = 0; k < EIN_; ++k) {
    const float av = a[k];
    const float4 w = *(const float4*)(eW + (size_t)k * D_ + c);
    acc.x += av * w.x; acc.y += av * w.y; acc.z += av * w.z; acc.w += av * w.w;
  }
  const float4 hv = *(const float4*)(h + (size_t)s * D_ + c);
  const float m0 = fmaxf(hv.x + acc.x, 0.0f);
  const float m1 = fmaxf(hv.y + acc.y, 0.0f);
  const float m2 = fmaxf(hv.z + acc.z, 0.0f);
  const float m3 = fmaxf(hv.w + acc.w, 0.0f);
  float* zp = zin + (size_t)d * D_ + c;
  atomicAdd(zp + 0, m0); atomicAdd(zp + 1, m1);
  atomicAdd(zp + 2, m2); atomicAdd(zp + 3, m3);
}

// Column sums / sum-of-squares over M rows (grid.y row-chunks, atomics)
__global__ void gnn_bn_stats(const float* __restrict__ X, float* __restrict__ stats,
                             int M, int N, int rowsPerBlock)
{
  const int col = blockIdx.x * blockDim.x + threadIdx.x; // N multiple of 256
  int r0 = blockIdx.y * rowsPerBlock;
  int r1 = r0 + rowsPerBlock; if (r1 > M) r1 = M;
  float s = 0.0f, s2 = 0.0f;
  for (int r = r0; r < r1; ++r) {
    const float v = X[(size_t)r * (size_t)N + col];
    s += v; s2 += v * v;
  }
  atomicAdd(&stats[col], s);
  atomicAdd(&stats[N + col], s2);
}

__global__ void gnn_bn_finalize(const float* __restrict__ stats, const float* __restrict__ g,
                                const float* __restrict__ b, float* __restrict__ ss,
                                int N, float invM)
{
  const int c = blockIdx.x * blockDim.x + threadIdx.x;
  if (c >= N) return;
  const float mean = stats[c] * invM;
  const float var  = stats[N + c] * invM - mean * mean;
  const float sc   = g[c] * rsqrtf(var + 1e-5f);
  ss[c]     = sc;
  ss[N + c] = b[c] - mean * sc;
}

// y = x*scale + shift; optional relu; writes f32 (outF) and/or bf16 (outB)
__global__ void gnn_bn_apply(const float* __restrict__ X, const float* __restrict__ ss,
                             float* __restrict__ outF, uint16_t* __restrict__ outB,
                             size_t total4, int nmask, int relu)
{
  size_t i = (size_t)blockIdx.x * blockDim.x + threadIdx.x;
  if (i >= total4) return;
  const size_t base = i << 2;
  const int c = (int)(base & (size_t)nmask);
  const int N = nmask + 1;
  const float4 x  = *(const float4*)(X + base);
  const float4 sc = *(const float4*)(ss + c);
  const float4 sh = *(const float4*)(ss + N + c);
  float y0 = x.x * sc.x + sh.x;
  float y1 = x.y * sc.y + sh.y;
  float y2 = x.z * sc.z + sh.z;
  float y3 = x.w * sc.w + sh.w;
  if (relu) {
    y0 = fmaxf(y0, 0.0f); y1 = fmaxf(y1, 0.0f);
    y2 = fmaxf(y2, 0.0f); y3 = fmaxf(y3, 0.0f);
  }
  if (outF) {
    float4 y; y.x = y0; y.y = y1; y.z = y2; y.w = y3;
    *(float4*)(outF + base) = y;
  }
  if (outB) {
    uint2 pv;
    pv.x = (uint32_t)f2bf(y0) | ((uint32_t)f2bf(y1) << 16);
    pv.y = (uint32_t)f2bf(y2) | ((uint32_t)f2bf(y3) << 16);
    *(uint2*)(outB + base) = pv;
  }
}

__global__ void gnn_count_nodes(const int* __restrict__ batch, float* __restrict__ counts, int n)
{
  int i = blockIdx.x * blockDim.x + threadIdx.x;
  if (i < n) atomicAdd(&counts[batch[i]], 1.0f);
}

__global__ void gnn_pool_sum(const float* __restrict__ h, const int* __restrict__ batch,
                             float* __restrict__ hg, size_t total4)
{
  size_t i = (size_t)blockIdx.x * blockDim.x + threadIdx.x;
  if (i >= total4) return;
  const int row = (int)(i >> 7);
  const int c4  = ((int)i & 127) << 2;
  const int b   = batch[row];
  const float4 v = *(const float4*)(h + (size_t)row * D_ + c4);
  float* hp = hg + (size_t)b * D_ + c4;
  atomicAdd(hp + 0, v.x); atomicAdd(hp + 1, v.y);
  atomicAdd(hp + 2, v.z); atomicAdd(hp + 3, v.w);
}

__global__ void gnn_pool_mean_cvt(const float* __restrict__ hg, const float* __restrict__ counts,
                                  uint16_t* __restrict__ out, int n)
{
  int i = blockIdx.x * blockDim.x + threadIdx.x;
  if (i >= n) return;
  const float c = fmaxf(counts[i >> 9], 1.0f);   // D_ = 512
  out[i] = f2bf(hg[i] / c);
}

// ---------------------------------------------------------------------------

extern "C" void kernel_launch(void* const* d_in, const int* in_sizes, int n_in,
                              void* d_out, int out_size, void* d_ws, size_t ws_size,
                              hipStream_t stream)
{
  (void)in_sizes; (void)n_in; (void)out_size; (void)ws_size;

  const float* x         = (const float*)d_in[0];
  const float* edge_attr = (const float*)d_in[1];
  const int*   eidx      = (const int*)d_in[2];
  const int*   batch     = (const int*)d_in[3];
  const float* node_W    = (const float*)d_in[4];
  const float* node_b    = (const float*)d_in[5];
  const float* eps       = (const float*)d_in[6];
  const float* edge_W    = (const float*)d_in[7];
  const float* edge_b    = (const float*)d_in[8];
  const float* mlp_W1    = (const float*)d_in[9];
  const float* mlp_b1    = (const float*)d_in[10];
  const float* bn1_g     = (const float*)d_in[11];
  const float* bn1_b     = (const float*)d_in[12];
  const float* mlp_W2    = (const float*)d_in[13];
  const float* mlp_b2    = (const float*)d_in[14];
  const float* bn_g      = (const float*)d_in[15];
  const float* bn_b      = (const float*)d_in[16];
  const float* vn_emb    = (const float*)d_in[17];
  const float* vn_W1     = (const float*)d_in[18];
  const float* vn_b1     = (const float*)d_in[19];
  const float* vn_g1     = (const float*)d_in[20];
  const float* vn_be1    = (const float*)d_in[21];
  const float* vn_W2     = (const float*)d_in[22];
  const float* vn_b2     = (const float*)d_in[23];
  const float* vn_g2     = (const float*)d_in[24];
  const float* vn_be2    = (const float*)d_in[25];
  const float* pred_W    = (const float*)d_in[26];
  const float* pred_b    = (const float*)d_in[27];

  const int* src = eidx;
  const int* dst = eidx + E_;

  // ---- workspace layout ----
  char* ws = (char*)d_ws;
  size_t off = 0;
  auto take = [&](size_t bytes) -> void* {
    void* p = ws + off;
    off += (bytes + 255) & ~(size_t)255;
    return p;
  };
  float*    h      = (float*)take((size_t)N_ * D_ * 4);
  float*    zin    = (float*)take((size_t)N_ * D_ * 4);
  float*    zh     = (float*)take((size_t)N_ * 2 * D_ * 4);
  uint16_t* abf    = (uint16_t*)take((size_t)N_ * 2 * D_ * 2);
  uint32_t* wpack  = (uint32_t*)take((size_t)4784128 * 4);
  float*    vn     = (float*)take((size_t)G_ * D_ * 4);
  float*    pooled = (float*)take((size_t)G_ * D_ * 4);
  float*    vnt    = (float*)take((size_t)G_ * 2 * D_ * 4);
  float*    hg     = (float*)take((size_t)G_ * D_ * 4);
  float*    counts = (float*)take((size_t)G_ * 4);
  float*    stats  = (float*)take((size_t)2 * 1024 * 4);
  float*    ssb    = (float*)take((size_t)2 * 1024 * 4);

  uint32_t* wp_node = wpack;                     // 64*512
  uint32_t* wp_W1   = wp_node + 32768;           // 5 * 256*1024
  uint32_t* wp_W2   = wp_W1 + 5 * 262144;        // 5 * 512*512
  uint32_t* wp_vn1  = wp_W2 + 5 * 262144;        // 4 * 256*1024
  uint32_t* wp_vn2  = wp_vn1 + 4 * 262144;       // 4 * 512*512
  uint32_t* wp_pred = wp_vn2 + 4 * 262144;       // 256*128

  auto packW = [&](const float* W, uint32_t* out, int K, int Nc) {
    size_t tot = ((size_t)(K >> 1)) * (size_t)Nc;
    gnn_pack_w<<<dim3((unsigned)((tot + 255) / 256)), dim3(256), 0, stream>>>(W, out, K, Nc);
  };
  auto gemm = [&](const uint16_t* A, const uint32_t* Bp, const float* bias, float* C,
                  int M, int K, int Nc) {
    dim3 grid(Nc / 64, M / 64), block(32, 2);
    gnn_wmma_gemm<<<grid, block, 0, stream>>>(A, Bp, bias, C, M, K, Nc);
  };
  auto cvt = [&](const float* in, uint16_t* out, size_t n) {
    size_t n2 = n >> 1;
    gnn_cvt_bf16<<<dim3((unsigned)((n2 + 255) / 256)), dim3(256), 0, stream>>>(in, out, n2);
  };
  auto bn = [&](const float* X, const float* g, const float* b, int M, int Nc,
                float* outF, uint16_t* outB, int relu) {
    gnn_zero_f32<<<dim3((2 * Nc + 255) / 256), dim3(256), 0, stream>>>(stats, (size_t)(2 * Nc));
    dim3 sg(Nc / 256, (M + 127) / 128);
    gnn_bn_stats<<<sg, dim3(256), 0, stream>>>(X, stats, M, Nc, 128);
    gnn_bn_finalize<<<dim3((Nc + 255) / 256), dim3(256), 0, stream>>>(stats, g, b, ssb, Nc, 1.0f / (float)M);
    size_t t4 = (size_t)M * (size_t)Nc / 4;
    gnn_bn_apply<<<dim3((unsigned)((t4 + 255) / 256)), dim3(256), 0, stream>>>(
        X, ssb, outF, outB, t4, Nc - 1, relu);
  };

  // ---- pack all weights to fragment-contiguous bf16 layout ----
  packW(node_W, wp_node, DIN_, D_);
  for (int l = 0; l < L_; ++l) {
    packW(mlp_W1 + (size_t)l * D_ * 2 * D_, wp_W1 + (size_t)l * 262144, D_, 2 * D_);
    packW(mlp_W2 + (size_t)l * 2 * D_ * D_, wp_W2 + (size_t)l * 262144, 2 * D_, D_);
  }
  for (int l = 0; l < L_ - 1; ++l) {
    packW(vn_W1 + (size_t)l * D_ * 2 * D_, wp_vn1 + (size_t)l * 262144, D_, 2 * D_);
    packW(vn_W2 + (size_t)l * 2 * D_ * D_, wp_vn2 + (size_t)l * 262144, 2 * D_, D_);
  }
  packW(pred_W, wp_pred, D_, T_);

  // ---- node encoder: h = x @ node_W + node_b ----
  cvt(x, abf, (size_t)N_ * DIN_);
  gemm(abf, wp_node, node_b, h, N_, DIN_, D_);

  // ---- virtual node init ----
  gnn_vn_init<<<dim3((G_ * D_ + 255) / 256), dim3(256), 0, stream>>>(vn, vn_emb, G_ * D_);

  const size_t nd4 = (size_t)N_ * (D_ / 4);

  for (int l = 0; l < L_; ++l) {
    const int doPool = (l < L_ - 1);
    if (doPool)
      gnn_copy_f32<<<dim3((G_ * D_ + 255) / 256), dim3(256), 0, stream>>>(pooled, vn, (size_t)G_ * D_);

    // h += vn[batch]; zin = (1+eps)*h; pooled += h
    gnn_add_vn<<<dim3((unsigned)((nd4 + 255) / 256)), dim3(256), 0, stream>>>(
        h, vn, batch, eps + l, zin, pooled, doPool);

    // zin += relu(h[src] + edge_attr @ edge_W[l] + edge_b[l]) scattered by dst
    gnn_edge_msg<<<dim3(E_), dim3(128), 0, stream>>>(
        h, edge_attr, src, dst, edge_W + (size_t)l * EIN_ * D_, edge_b + (size_t)l * D_, zin);

    // node MLP: GEMM1 -> BN1+relu -> GEMM2 -> BN2(+relu) -> h
    cvt(zin, abf, (size_t)N_ * D_);
    gemm(abf, wp_W1 + (size_t)l * 262144, mlp_b1 + (size_t)l * 2 * D_, zh, N_, D_, 2 * D_);
    bn(zh, bn1_g + (size_t)l * 2 * D_, bn1_b + (size_t)l * 2 * D_, N_, 2 * D_, nullptr, abf, 1);
    gemm(abf, wp_W2 + (size_t)l * 262144, mlp_b2 + (size_t)l * D_, zin, N_, 2 * D_, D_);
    bn(zin, bn_g + (size_t)l * D_, bn_b + (size_t)l * D_, N_, D_, h, nullptr, doPool);

    if (doPool) {
      // virtual-node MLP update
      cvt(pooled, abf, (size_t)G_ * D_);
      gemm(abf, wp_vn1 + (size_t)l * 262144, vn_b1 + (size_t)l * 2 * D_, vnt, G_, D_, 2 * D_);
      bn(vnt, vn_g1 + (size_t)l * 2 * D_, vn_be1 + (size_t)l * 2 * D_, G_, 2 * D_, nullptr, abf, 1);
      gemm(abf, wp_vn2 + (size_t)l * 262144, vn_b2 + (size_t)l * D_, pooled, G_, 2 * D_, D_);
      bn(pooled, vn_g2 + (size_t)l * D_, vn_be2 + (size_t)l * D_, G_, D_, vn, nullptr, 1);
    }
  }

  // ---- mean pool + prediction head ----
  gnn_zero_f32<<<dim3((G_ * D_ + 255) / 256), dim3(256), 0, stream>>>(hg, (size_t)G_ * D_);
  gnn_zero_f32<<<dim3((G_ + 255) / 256), dim3(256), 0, stream>>>(counts, (size_t)G_);
  gnn_count_nodes<<<dim3((N_ + 255) / 256), dim3(256), 0, stream>>>(batch, counts, N_);
  gnn_pool_sum<<<dim3((unsigned)((nd4 + 255) / 256)), dim3(256), 0, stream>>>(h, batch, hg, nd4);
  gnn_pool_mean_cvt<<<dim3((G_ * D_ + 255) / 256), dim3(256), 0, stream>>>(hg, counts, abf, G_ * D_);
  gemm(abf, wp_pred, pred_b, (float*)d_out, G_, D_, T_);
}
